// Attention_22454089023887
// MI455X (gfx1250) — matrix operand order repaired and checked by reading, not compiled, not run on hardware
//
#include <hip/hip_runtime.h>

// ---------------------------------------------------------------------------
// CDNA5 (gfx1250) fused LeViT-style attention block, wave32 + WMMA f16.
//  x -> LayerNorm -> QKV gemm -> flash attention (+rel-pos bias) -> proj gemm
// All matmuls use v_wmma_f32_16x16x32_f16. Scores never touch HBM.
// ---------------------------------------------------------------------------

typedef __attribute__((ext_vector_type(16))) _Float16 v16h;
typedef __attribute__((ext_vector_type(8)))  _Float16 v8h;
typedef __attribute__((ext_vector_type(8)))  float    v8f;

#define WMMA(A, B, C) \
  __builtin_amdgcn_wmma_f32_16x16x32_f16(false, (A), false, (B), (short)0, (C), false, false)

static constexpr int   BB   = 32;
static constexpr int   NN   = 1024;
static constexpr int   DIMC = 256;
static constexpr int   HH   = 8;
static constexpr int   KDQ  = 16;   // qk head dim
static constexpr int   DV   = 64;   // v head dim
static constexpr int   DHX  = 512;  // H*DV
static constexpr int   QKVC = 768;  // 2*KD*H + DH
static constexpr float SCALE = 0.25f;  // KD^-0.5
static constexpr float EPS   = 1e-5f;

__device__ inline v8h v8h_zero() {
  v8h z = {(_Float16)0,(_Float16)0,(_Float16)0,(_Float16)0,
           (_Float16)0,(_Float16)0,(_Float16)0,(_Float16)0};
  return z;
}
__device__ inline v8f v8f_zero() {
  v8f z = {0.f,0.f,0.f,0.f,0.f,0.f,0.f,0.f};
  return z;
}

// ---- WMMA fragment loaders (wave32 layouts per CDNA5 ISA 7.12.2) ----------
// A (16x32 f16): lane L -> row M = L%16; halves t=0..7 -> K = 8*(L/16)+t,
//                halves t=8..15 -> K = 16 + 8*(L/16) + (t-8).
__device__ inline v16h load_a_frag(const _Float16* __restrict__ A, int lda,
                                   int m0, int k0) {
  int lane = threadIdx.x & 31, g = lane >> 4, row = m0 + (lane & 15);
  const _Float16* p = A + (long)row * lda + k0;
  union { v16h v; v8h h[2]; } u;
  u.h[0] = *(const v8h*)(p + 8 * g);
  u.h[1] = *(const v8h*)(p + 16 + 8 * g);
  return u.v;
}
// A with K=16 zero-padded to 32 (for Q·K^T, KD=16): halves 8..15 are K>=16 -> 0.
__device__ inline v16h load_a_frag_k16(const _Float16* __restrict__ A, int m0) {
  int lane = threadIdx.x & 31, g = lane >> 4, row = m0 + (lane & 15);
  union { v16h v; v8h h[2]; } u;
  u.h[0] = *(const v8h*)(A + (long)row * KDQ + 8 * g);
  u.h[1] = v8h_zero();
  return u.v;
}
// A fragment from an LDS tile (used for the P matrix in P·V).
__device__ inline v16h load_a_frag_lds(const _Float16* P, int lda, int k0) {
  int lane = threadIdx.x & 31, g = lane >> 4, row = lane & 15;
  const _Float16* p = P + row * lda + k0;
  union { v16h v; v8h h[2]; } u;
  u.h[0] = *(const v8h*)(p + 8 * g);
  u.h[1] = *(const v8h*)(p + 16 + 8 * g);
  return u.v;
}
// B (32x16 f16): lane L -> col N = L%16; halves t=0..7 -> K = 8*(L/16)+t,
// halves t=8..15 -> K = 16 + 8*(L/16) + (t-8). Bw is stored row-major (N x K).
__device__ inline v16h load_b_frag(const _Float16* __restrict__ Bw, int ldb,
                                   int n0, int k0) {
  int lane = threadIdx.x & 31, g = lane >> 4, col = n0 + (lane & 15);
  const _Float16* p = Bw + (long)col * ldb + k0;
  union { v16h v; v8h h[2]; } u;
  u.h[0] = *(const v8h*)(p + 8 * g);
  u.h[1] = *(const v8h*)(p + 16 + 8 * g);
  return u.v;
}
// B with K=16 zero-padded to 32: halves 8..15 are K>=16 -> 0.
__device__ inline v16h load_b_frag_k16(const _Float16* __restrict__ Kt, int n0) {
  int lane = threadIdx.x & 31, g = lane >> 4, col = n0 + (lane & 15);
  union { v16h v; v8h h[2]; } u;
  u.h[0] = *(const v8h*)(Kt + (long)col * KDQ + 8 * g);
  u.h[1] = v8h_zero();
  return u.v;
}

// ---------------------------------------------------------------------------
// Kernel 1: elementwise f32 -> f16 cast (weights)
// ---------------------------------------------------------------------------
__global__ void cast_f32_to_f16(const float* __restrict__ in,
                                _Float16* __restrict__ out, int n) {
  for (int i = blockIdx.x * blockDim.x + threadIdx.x; i < n;
       i += gridDim.x * blockDim.x)
    out[i] = (_Float16)in[i];
}

// ---------------------------------------------------------------------------
// Kernel 2: LayerNorm(x) * gamma + beta  -> f16.  One wave per 256-wide row.
// ---------------------------------------------------------------------------
__global__ __launch_bounds__(256) void ln_cast_kernel(
    const float* __restrict__ x, const float* __restrict__ gamma,
    const float* __restrict__ beta, _Float16* __restrict__ xn) {
  int wave = threadIdx.x >> 5, lane = threadIdx.x & 31;
  long row = (long)blockIdx.x * 8 + wave;
  const float* xr = x + row * DIMC + lane * 8;
  float4 a0 = *(const float4*)(xr);
  float4 a1 = *(const float4*)(xr + 4);
  float s = a0.x + a0.y + a0.z + a0.w + a1.x + a1.y + a1.z + a1.w;
#pragma unroll
  for (int m = 1; m < 32; m <<= 1) s += __shfl_xor(s, m, 32);
  float mu = s * (1.0f / DIMC);
  float d0x = a0.x - mu, d0y = a0.y - mu, d0z = a0.z - mu, d0w = a0.w - mu;
  float d1x = a1.x - mu, d1y = a1.y - mu, d1z = a1.z - mu, d1w = a1.w - mu;
  float vs = d0x * d0x + d0y * d0y + d0z * d0z + d0w * d0w +
             d1x * d1x + d1y * d1y + d1z * d1z + d1w * d1w;
#pragma unroll
  for (int m = 1; m < 32; m <<= 1) vs += __shfl_xor(vs, m, 32);
  float rs = rsqrtf(vs * (1.0f / DIMC) + EPS);
  const float4 g0 = *(const float4*)(gamma + lane * 8);
  const float4 g1 = *(const float4*)(gamma + lane * 8 + 4);
  const float4 b0 = *(const float4*)(beta + lane * 8);
  const float4 b1 = *(const float4*)(beta + lane * 8 + 4);
  union { v8h v; _Float16 e[8]; } o;
  o.e[0] = (_Float16)(d0x * rs * g0.x + b0.x);
  o.e[1] = (_Float16)(d0y * rs * g0.y + b0.y);
  o.e[2] = (_Float16)(d0z * rs * g0.z + b0.z);
  o.e[3] = (_Float16)(d0w * rs * g0.w + b0.w);
  o.e[4] = (_Float16)(d1x * rs * g1.x + b1.x);
  o.e[5] = (_Float16)(d1y * rs * g1.y + b1.y);
  o.e[6] = (_Float16)(d1z * rs * g1.z + b1.z);
  o.e[7] = (_Float16)(d1w * rs * g1.w + b1.w);
  *(v8h*)(xn + row * DIMC + lane * 8) = o.v;
}

// ---------------------------------------------------------------------------
// Kernel 3: QKV GEMM  (BN x 256) x (768 x 256)^T  via WMMA, scatter epilogue.
// Block = 256 threads (8 waves), block tile 128x64, wave tile 32x32.
// q scaled by SCALE; K stored (b,h,n,16); V stored transposed (b,h,d,n).
// ---------------------------------------------------------------------------
__global__ __launch_bounds__(256) void qkv_gemm_kernel(
    const _Float16* __restrict__ Ah, const _Float16* __restrict__ Wh,
    const float* __restrict__ bqkv, _Float16* __restrict__ qbuf,
    _Float16* __restrict__ kbuf, _Float16* __restrict__ vtbuf) {
  int w = threadIdx.x >> 5;
  int m = blockIdx.x * 128 + (w & 3) * 32;
  int n = blockIdx.y * 64 + (w >> 2) * 32;
  v8f acc[2][2];
#pragma unroll
  for (int i = 0; i < 2; ++i)
#pragma unroll
    for (int j = 0; j < 2; ++j) acc[i][j] = v8f_zero();

  for (int k0 = 0; k0 < DIMC; k0 += 32) {
    if (k0 + 32 < DIMC)
      __builtin_prefetch(Ah + (long)(m + (threadIdx.x & 15)) * DIMC + k0 + 32, 0, 1);
    v16h a0 = load_a_frag(Ah, DIMC, m, k0);
    v16h a1 = load_a_frag(Ah, DIMC, m + 16, k0);
    v16h b0 = load_b_frag(Wh, DIMC, n, k0);
    v16h b1 = load_b_frag(Wh, DIMC, n + 16, k0);
    acc[0][0] = WMMA(a0, b0, acc[0][0]);
    acc[0][1] = WMMA(a0, b1, acc[0][1]);
    acc[1][0] = WMMA(a1, b0, acc[1][0]);
    acc[1][1] = WMMA(a1, b1, acc[1][1]);
  }
  int lane = threadIdx.x & 31, g = lane >> 4, l15 = lane & 15;
#pragma unroll
  for (int i = 0; i < 2; ++i)
#pragma unroll
    for (int j = 0; j < 2; ++j)
#pragma unroll
      for (int r = 0; r < 8; ++r) {
        int mr = m + 16 * i + r + 8 * g;       // global row (b*N + n)
        int c  = n + 16 * j + l15;             // output channel 0..767
        float v = acc[i][j][r] + bqkv[c];
        int bidx = mr >> 10, nn = mr & 1023;
        int h = c / 96, r96 = c % 96;
        long bh = (long)bidx * HH + h;
        if (r96 < 16)
          qbuf[(bh * NN + nn) * KDQ + r96] = (_Float16)(v * SCALE);
        else if (r96 < 32)
          kbuf[(bh * NN + nn) * KDQ + (r96 - 16)] = (_Float16)v;
        else
          vtbuf[(bh * DV + (r96 - 32)) * NN + nn] = (_Float16)v;
      }
}

// ---------------------------------------------------------------------------
// Kernel 4: flash attention per (b,h).  Block = 128 threads (4 waves); each
// wave owns 16 query rows; loop over 16 key chunks of 64.  Online softmax
// with per-row stats held redundantly across the 16 lanes of a half-wave.
// P is transposed C-layout -> A-layout through a private LDS tile.
// ---------------------------------------------------------------------------
__global__ __launch_bounds__(128) void attn_kernel(
    const _Float16* __restrict__ qbuf, const _Float16* __restrict__ kbuf,
    const _Float16* __restrict__ vtbuf, const float* __restrict__ biases,
    const int* __restrict__ idxs, _Float16* __restrict__ aout) {
  int w = threadIdx.x >> 5, lane = threadIdx.x & 31;
  int g = lane >> 4, l15 = lane & 15;
  int h = blockIdx.y, b = blockIdx.z;
  long bh = (long)b * HH + h;
  const _Float16* Q  = qbuf + bh * NN * KDQ;
  const _Float16* Kt = kbuf + bh * NN * KDQ;
  const _Float16* Vt = vtbuf + bh * DV * NN;
  const float* biasH = biases + h * NN;
  int q0 = blockIdx.x * 64;
  int rowbase = q0 + w * 16;

  __shared__ _Float16 Psh[4][16][72];  // per-wave 16x64 P tile (+pad)

  v16h aq = load_a_frag_k16(Q, rowbase);  // SCALE already folded into q
  v8f o[4];
#pragma unroll
  for (int nt = 0; nt < 4; ++nt) o[nt] = v8f_zero();
  float mi[8], li[8];
#pragma unroll
  for (int r = 0; r < 8; ++r) { mi[r] = -1e30f; li[r] = 0.f; }

  for (int j = 0; j < 16; ++j) {
    int kbase = j * 64;
    // ---- scores S = q @ k^T (SCALE folded) -------------------------------
    v8f s[4];
#pragma unroll
    for (int nt = 0; nt < 4; ++nt) {
      v16h bk = load_b_frag_k16(Kt, kbase + nt * 16);
      s[nt] = WMMA(aq, bk, v8f_zero());
    }
    // ---- + relative position bias (gather, L2-resident tables) ----------
#pragma unroll
    for (int nt = 0; nt < 4; ++nt) {
      int jc = kbase + nt * 16 + l15;
#pragma unroll
      for (int r = 0; r < 8; ++r) {
        int row = rowbase + r + 8 * g;
        s[nt][r] += biasH[idxs[row * NN + jc]];
      }
    }
    // ---- online softmax over this 64-key chunk ---------------------------
#pragma unroll
    for (int r = 0; r < 8; ++r) {
      float mx = fmaxf(fmaxf(s[0][r], s[1][r]), fmaxf(s[2][r], s[3][r]));
#pragma unroll
      for (int msk = 1; msk < 16; msk <<= 1) mx = fmaxf(mx, __shfl_xor(mx, msk, 16));
      float mn = fmaxf(mi[r], mx);
      float corr = __expf(mi[r] - mn);
      float ls = 0.f;
#pragma unroll
      for (int nt = 0; nt < 4; ++nt) {
        float p = __expf(s[nt][r] - mn);
        s[nt][r] = p;
        ls += p;
      }
#pragma unroll
      for (int msk = 1; msk < 16; msk <<= 1) ls += __shfl_xor(ls, msk, 16);
      li[r] = li[r] * corr + ls;
      mi[r] = mn;
#pragma unroll
      for (int nt = 0; nt < 4; ++nt) o[nt][r] *= corr;
    }
    // ---- transpose P into A-fragment layout via private LDS tile ---------
    // (intra-wave only: DS ops from one wave complete in order, no barrier)
#pragma unroll
    for (int nt = 0; nt < 4; ++nt)
#pragma unroll
      for (int r = 0; r < 8; ++r)
        Psh[w][r + 8 * g][nt * 16 + l15] = (_Float16)s[nt][r];
    // ---- O += P @ V  (V pre-transposed so B-frag loads are contiguous) ---
#pragma unroll
    for (int ks = 0; ks < 2; ++ks) {
      v16h ap = load_a_frag_lds(&Psh[w][0][0], 72, ks * 32);
#pragma unroll
      for (int nt = 0; nt < 4; ++nt) {
        v16h bv = load_b_frag(Vt, NN, nt * 16, kbase + ks * 32);
        o[nt] = WMMA(ap, bv, o[nt]);
      }
    }
  }
  // ---- epilogue: O / l, write (b, n, h*64+d) f16 --------------------------
#pragma unroll
  for (int nt = 0; nt < 4; ++nt)
#pragma unroll
    for (int r = 0; r < 8; ++r) {
      int row = rowbase + r + 8 * g;
      int d = nt * 16 + l15;
      float val = o[nt][r] * (1.0f / li[r]);
      aout[((long)b * NN + row) * DHX + h * DV + d] = (_Float16)val;
    }
}

// ---------------------------------------------------------------------------
// Kernel 5: output projection  (BN x 512) x (256 x 512)^T + b_proj  -> f32
// ---------------------------------------------------------------------------
__global__ __launch_bounds__(256) void proj_gemm_kernel(
    const _Float16* __restrict__ Ah, const _Float16* __restrict__ Wh,
    const float* __restrict__ bproj, float* __restrict__ out) {
  int w = threadIdx.x >> 5;
  int m = blockIdx.x * 128 + (w & 3) * 32;
  int n = blockIdx.y * 64 + (w >> 2) * 32;
  v8f acc[2][2];
#pragma unroll
  for (int i = 0; i < 2; ++i)
#pragma unroll
    for (int j = 0; j < 2; ++j) acc[i][j] = v8f_zero();

  for (int k0 = 0; k0 < DHX; k0 += 32) {
    if (k0 + 32 < DHX)
      __builtin_prefetch(Ah + (long)(m + (threadIdx.x & 15)) * DHX + k0 + 32, 0, 1);
    v16h a0 = load_a_frag(Ah, DHX, m, k0);
    v16h a1 = load_a_frag(Ah, DHX, m + 16, k0);
    v16h b0 = load_b_frag(Wh, DHX, n, k0);
    v16h b1 = load_b_frag(Wh, DHX, n + 16, k0);
    acc[0][0] = WMMA(a0, b0, acc[0][0]);
    acc[0][1] = WMMA(a0, b1, acc[0][1]);
    acc[1][0] = WMMA(a1, b0, acc[1][0]);
    acc[1][1] = WMMA(a1, b1, acc[1][1]);
  }
  int lane = threadIdx.x & 31, g = lane >> 4, l15 = lane & 15;
#pragma unroll
  for (int i = 0; i < 2; ++i)
#pragma unroll
    for (int j = 0; j < 2; ++j)
#pragma unroll
      for (int r = 0; r < 8; ++r) {
        long mr = m + 16 * i + r + 8 * g;
        int c = n + 16 * j + l15;
        out[mr * DIMC + c] = acc[i][j][r] + bproj[c];
      }
}

// ---------------------------------------------------------------------------
extern "C" void kernel_launch(void* const* d_in, const int* in_sizes, int n_in,
                              void* d_out, int out_size, void* d_ws,
                              size_t ws_size, hipStream_t stream) {
  (void)in_sizes; (void)n_in; (void)out_size; (void)ws_size;
  const float* x        = (const float*)d_in[0];
  const float* gamma    = (const float*)d_in[1];
  const float* beta     = (const float*)d_in[2];
  const float* w_qkv    = (const float*)d_in[3];
  const float* b_qkv    = (const float*)d_in[4];
  const float* w_proj   = (const float*)d_in[5];
  const float* b_proj   = (const float*)d_in[6];
  const float* biases   = (const float*)d_in[7];
  const int*   bias_idx = (const int*)d_in[8];
  float* out = (float*)d_out;

  char* ws = (char*)d_ws;
  const long BN = (long)BB * NN;
  size_t off = 0;
  _Float16* xn_h  = (_Float16*)(ws + off); off += BN * DIMC * 2;          // 16 MB
  _Float16* wqk_h = (_Float16*)(ws + off); off += (size_t)QKVC * DIMC * 2;// 384 KB
  _Float16* wpr_h = (_Float16*)(ws + off); off += (size_t)DIMC * DHX * 2; // 256 KB
  _Float16* qbuf  = (_Float16*)(ws + off); off += (size_t)BB*HH*NN*KDQ*2; // 8 MB
  _Float16* kbuf  = (_Float16*)(ws + off); off += (size_t)BB*HH*NN*KDQ*2; // 8 MB
  _Float16* vtbuf = (_Float16*)(ws + off); off += (size_t)BB*HH*DV*NN*2;  // 32 MB
  _Float16* aout  = (_Float16*)(ws + off); off += BN * DHX * 2;           // 32 MB

  // 1) weight casts to f16
  cast_f32_to_f16<<<256, 256, 0, stream>>>(w_qkv, wqk_h, QKVC * DIMC);
  cast_f32_to_f16<<<256, 256, 0, stream>>>(w_proj, wpr_h, DIMC * DHX);
  // 2) LayerNorm + cast (one wave per row, 8 rows per block)
  ln_cast_kernel<<<BN / 8, 256, 0, stream>>>(x, gamma, beta, xn_h);
  // 3) QKV GEMM + scatter into q (scaled) / k / v^T layouts
  qkv_gemm_kernel<<<dim3(BN / 128, QKVC / 64), 256, 0, stream>>>(
      xn_h, wqk_h, b_qkv, qbuf, kbuf, vtbuf);
  // 4) flash attention with relative-position bias
  attn_kernel<<<dim3(NN / 64, HH, BB), 128, 0, stream>>>(
      qbuf, kbuf, vtbuf, biases, bias_idx, aout);
  // 5) output projection
  proj_gemm_kernel<<<dim3(BN / 128, DIMC / 64), 256, 0, stream>>>(
      aout, wpr_h, b_proj, out);
}
